// SinkhornMproto_60670708023301
// MI455X (gfx1250) — matrix operand order repaired and checked by reading, not compiled, not run on hardware
//
#include <hip/hip_runtime.h>

// MI455X / gfx1250, wave32. WMMA fragment types.
typedef float    v2f  __attribute__((ext_vector_type(2)));
typedef float    v8f  __attribute__((ext_vector_type(8)));
typedef _Float16 v16h __attribute__((ext_vector_type(16)));
typedef _Float16 v8h  __attribute__((ext_vector_type(8)));

#define SK_EPS 0.001f

__device__ __forceinline__ float fast_rcp(float x) { return __builtin_amdgcn_rcpf(x); }
__device__ __forceinline__ float relu(float x) { return x > 0.0f ? x : 0.0f; }

// One wave handles 32 rows: 2 M-tiles of 16 through the MLP (WMMA),
// then 1 row per lane for exp + 10 Sinkhorn iterations.
__global__ __launch_bounds__(256) void sinkhorn_mlp_kernel(
    const float* __restrict__ margins,
    const float* __restrict__ W1, const float* __restrict__ b1,
    const float* __restrict__ W2, const float* __restrict__ b2,
    const float* __restrict__ W3, const float* __restrict__ b3,
    const float* __restrict__ W4, const float* __restrict__ b4,
    float* __restrict__ out, int nrows)
{
    // Per-wave private staging: 2KB f16 activations + 1KB f32 pars (+1KB pad)
    __shared__ __align__(16) char smem_raw[8 * 4096];
    const int lane = threadIdx.x & 31;
    const int wave = threadIdx.x >> 5;
    const int hs   = lane >> 4;     // half-select (lanes 16-31)
    const int l    = lane & 15;
    _Float16* lds_h = (_Float16*)(smem_raw + wave * 4096);
    float*    lds_p = (float*)   (smem_raw + wave * 4096 + 2048);

    const long rowBase = (long)(blockIdx.x * 8 + wave) * 32;
    const long lastRow = (long)nrows - 1;

    // ---------------- Weight fragments (uniform, built once) ----------------
    // Layer1: W1 (64,4). B[k][n] = W1[n][k]. f32 B 4x16 frag: lane<16 holds
    // {K0,K1}, lane>=16 holds {K2,K3} for column N = 16t+l.
    v2f wb1[4]; float bias1[4];
    #pragma unroll
    for (int t = 0; t < 4; ++t) {
        int n = 16 * t + l;
        wb1[t].x = W1[n * 4 + hs * 2];
        wb1[t].y = W1[n * 4 + hs * 2 + 1];
        bias1[t] = b1[n];
    }
    // Layer2: W2 (32,64). f16 B 32x16 frag: elems 0-7 -> K hs*8+0..7,
    // elems 8-15 -> K hs*8+16..23 (within each K-chunk of 32) for N = 16t+l.
    v16h wb2[2][2]; float bias2[2];
    #pragma unroll
    for (int t = 0; t < 2; ++t) {
        int n = 16 * t + l;
        bias2[t] = b2[n];
        #pragma unroll
        for (int kc = 0; kc < 2; ++kc) {
            int kb = 32 * kc + hs * 8;
            #pragma unroll
            for (int h = 0; h < 16; ++h)
                wb2[t][kc][h] = (_Float16)W2[n * 64 + kb + h + (h & 8)];
        }
    }
    // Layer3: W3 (16,32), single 16x16x32 tile.
    v16h wb3; float bias3 = b3[l];
    #pragma unroll
    for (int h = 0; h < 16; ++h)
        wb3[h] = (_Float16)W3[l * 32 + hs * 8 + h + (h & 8)];
    // Layer4: W4 (7,16), K padded 16->32, N padded 7->16 with zeros.
    v16h wb4; float bias4 = (l < 7) ? b4[l] : 0.0f;
    #pragma unroll
    for (int h = 0; h < 16; ++h)
        wb4[h] = (h < 8 && l < 7) ? (_Float16)W4[l * 16 + hs * 8 + h] : (_Float16)0.0f;

    // ---------------- MLP: two 16-row M-tiles through 4 layers ----------------
    #pragma unroll
    for (int mt = 0; mt < 2; ++mt) {
        // Layer 1: A = margins tile, f32 A 16x4 layout: lane<16 {K0,K1}, lane>=16 {K2,K3}
        long m = rowBase + 16 * mt + l;
        if (m > lastRow) m = lastRow;           // clamp (keeps EXEC uniform for WMMA)
        v2f a1;
        a1.x = margins[m * 4 + hs * 2];
        a1.y = margins[m * 4 + hs * 2 + 1];

        #pragma unroll
        for (int t = 0; t < 4; ++t) {
            v8f c = {};
            c = __builtin_amdgcn_wmma_f32_16x16x4_f32(
                    false, a1, false, wb1[t], (short)0, c, false, false);
            // C layout: VGPR r, lane -> (M = r + hs*8, N = l). Bias+ReLU -> f16 LDS, stride 64.
            #pragma unroll
            for (int r = 0; r < 8; ++r) {
                float v = relu(c[r] + bias1[t]);
                lds_h[(hs * 8 + r) * 64 + 16 * t + l] = (_Float16)v;
            }
        }
        asm volatile("s_wait_dscnt 0x0" ::: "memory");

        // Layer 2: A frags from LDS (f16 A 16x32 layout: halves 0-7 -> K hs*8+0..7,
        // halves 8-15 -> K hs*8+16..23), two K-chunks of 32.
        v16h a2[2];
        #pragma unroll
        for (int kc = 0; kc < 2; ++kc) {
            const _Float16* base = lds_h + l * 64 + kc * 32 + hs * 8;
            v8h lo = *(const v8h*)(base);
            v8h hi = *(const v8h*)(base + 16);
            #pragma unroll
            for (int h = 0; h < 8; ++h) { a2[kc][h] = lo[h]; a2[kc][8 + h] = hi[h]; }
        }
        asm volatile("" ::: "memory");
        #pragma unroll
        for (int t = 0; t < 2; ++t) {
            v8f c = {};
            c = __builtin_amdgcn_wmma_f32_16x16x32_f16(
                    false, a2[0], false, wb2[t][0], (short)0, c, false, false);
            c = __builtin_amdgcn_wmma_f32_16x16x32_f16(
                    false, a2[1], false, wb2[t][1], (short)0, c, false, false);
            #pragma unroll
            for (int r = 0; r < 8; ++r) {
                float v = relu(c[r] + bias2[t]);
                lds_h[(hs * 8 + r) * 32 + 16 * t + l] = (_Float16)v;  // stride 32
            }
        }
        asm volatile("s_wait_dscnt 0x0" ::: "memory");

        // Layer 3: K=32 exactly, one WMMA.
        v16h a3;
        {
            const _Float16* base = lds_h + l * 32 + hs * 8;
            v8h lo = *(const v8h*)(base);
            v8h hi = *(const v8h*)(base + 16);
            #pragma unroll
            for (int h = 0; h < 8; ++h) { a3[h] = lo[h]; a3[8 + h] = hi[h]; }
        }
        asm volatile("" ::: "memory");
        v8f c3 = {};
        c3 = __builtin_amdgcn_wmma_f32_16x16x32_f16(
                 false, a3, false, wb3, (short)0, c3, false, false);
        #pragma unroll
        for (int r = 0; r < 8; ++r) {
            float v = relu(c3[r] + bias3);
            lds_h[(hs * 8 + r) * 16 + l] = (_Float16)v;               // stride 16
        }
        asm volatile("s_wait_dscnt 0x0" ::: "memory");

        // Layer 4: K=16 real (lane<16 -> K0..7, lane>=16 -> K8..15), upper half zero.
        v16h a4;
        {
            const _Float16* base = lds_h + l * 16 + hs * 8;
            v8h lo = *(const v8h*)(base);
            #pragma unroll
            for (int h = 0; h < 8; ++h) { a4[h] = lo[h]; a4[8 + h] = (_Float16)0.0f; }
        }
        asm volatile("" ::: "memory");
        v8f c4 = {};
        c4 = __builtin_amdgcn_wmma_f32_16x16x32_f16(
                 false, a4, false, wb4, (short)0, c4, false, false);

        // Transpose pars to one-row-per-lane layout (cols 0..6 used, stride 8).
        if (l < 8) {
            #pragma unroll
            for (int r = 0; r < 8; ++r)
                lds_p[(16 * mt + hs * 8 + r) * 8 + l] = c4[r] + bias4;
        }
    }
    asm volatile("s_wait_dscnt 0x0" ::: "memory");

    // ---------------- Per-row epilogue: exp + 10 Sinkhorn iterations ----------------
    long row = rowBase + lane;
    if (row < (long)nrows) {
        float p0 = lds_p[lane * 8 + 0], p1 = lds_p[lane * 8 + 1];
        float p2 = lds_p[lane * 8 + 2], p3 = lds_p[lane * 8 + 3];
        float p4 = lds_p[lane * 8 + 4], p5 = lds_p[lane * 8 + 5];
        float p6 = lds_p[lane * 8 + 6];
        float ep0 = __expf(p0), ep1 = __expf(p1), ep2 = __expf(p2);
        float ep3 = __expf(p3), ep4 = __expf(p4), ep5 = __expf(p5);
        float ep6 = __expf(p6);

        const float4 mg = *(const float4*)(margins + row * 4);

        // mu0 = clamp(muc, EPS); rowm = [ep2, ep3]; colm = [m0, m1]
        float mu00 = fmaxf(ep0, SK_EPS), mu01 = SK_EPS,            mu02 = fmaxf(ep2, SK_EPS);
        float mu10 = SK_EPS,             mu11 = fmaxf(ep1, SK_EPS), mu12 = fmaxf(ep3, SK_EPS);
        float mu20 = fmaxf(ep4, SK_EPS), mu21 = fmaxf(ep5, SK_EPS), mu22 = SK_EPS;
        const float rm0 = ep2, rm1 = ep3, cm0 = mg.x, cm1 = mg.y;

        #pragma unroll
        for (int it = 0; it < 10; ++it) {
            float f0 = rm0 * fast_rcp(mu00 + mu01 + mu02);
            float f1 = rm1 * fast_rcp(mu10 + mu11 + mu12);
            mu00 *= f0; mu01 *= f0; mu02 *= f0;
            mu10 *= f1; mu11 *= f1; mu12 *= f1;
            float g0 = cm0 * fast_rcp(mu00 + mu10 + mu20);
            float g1 = cm1 * fast_rcp(mu01 + mu11 + mu21);
            mu00 *= g0; mu10 *= g0; mu20 *= g0;
            mu01 *= g1; mu11 *= g1; mu21 *= g1;
        }

        float* o = out + row * 9;
        o[0] = mu00; o[1] = mu01; o[2] = mu02;
        o[3] = mu10; o[4] = mu11; o[5] = mu12;
        o[6] = mu20; o[7] = mu21; o[8] = mu22;
        out[(long)nrows * 9 + row] = ep6;   // V = exp(pars[:, -1])
    }
}

extern "C" void kernel_launch(void* const* d_in, const int* in_sizes, int n_in,
                              void* d_out, int out_size, void* d_ws, size_t ws_size,
                              hipStream_t stream) {
    const float* margins = (const float*)d_in[0];
    const float* W1 = (const float*)d_in[1]; const float* b1 = (const float*)d_in[2];
    const float* W2 = (const float*)d_in[3]; const float* b2 = (const float*)d_in[4];
    const float* W3 = (const float*)d_in[5]; const float* b3 = (const float*)d_in[6];
    const float* W4 = (const float*)d_in[7]; const float* b4 = (const float*)d_in[8];
    float* out = (float*)d_out;
    int nrows = in_sizes[0] / 4;
    if (nrows <= 0) return;
    int blocks = (nrows + 255) / 256;   // 256 rows per block (8 waves x 32 rows)
    sinkhorn_mlp_kernel<<<blocks, 256, 0, stream>>>(
        margins, W1, b1, W2, b2, W3, b3, W4, b4, out, nrows);
}
